// FlaxAttention_17583596110634
// MI455X (gfx1250) — matrix operand order repaired and checked by reading, not compiled, not run on hardware
//
#include <hip/hip_runtime.h>

typedef _Float16 half_t;
typedef __attribute__((ext_vector_type(16))) _Float16 v16h;
typedef __attribute__((ext_vector_type(8)))  _Float16 v8h;
typedef __attribute__((ext_vector_type(8)))  float    v8f;
typedef __attribute__((ext_vector_type(4)))  unsigned int v4u;
typedef __attribute__((ext_vector_type(8)))  int      v8i;
typedef __attribute__((ext_vector_type(4)))  int      v4i;

// ---------------------------------------------------------------------------
// helpers
// ---------------------------------------------------------------------------
__device__ __forceinline__ v16h combine16(v8h a, v8h b) {
  v16h r;
#pragma unroll
  for (int i = 0; i < 8; ++i) { r[i] = a[i]; r[i + 8] = b[i]; }
  return r;
}

__device__ __forceinline__ v8f wmma_f16(v16h a, v16h b, v8f c) {
  // D = A(16x32 f16) * B(32x16 f16) + C(16x16 f32)
  return __builtin_amdgcn_wmma_f32_16x16x32_f16(
      /*neg_a=*/false, a, /*neg_b=*/false, b,
      /*c_mod=*/(short)0, c, /*reuse_a=*/false, /*reuse_b=*/false);
}

// TDM: 2D tile (global -> LDS). D# per CDNA5 ISA 8.3/8.4.
// tensor viewed from the tile start; data_size = 2 bytes (f16).
__device__ __forceinline__ void tdm_load_2d(unsigned lds_addr, const void* gp,
                                            unsigned tensor_d0, unsigned tensor_d1,
                                            unsigned tile_d0, unsigned tile_d1,
                                            unsigned long long stride0) {
  unsigned long long ga = (unsigned long long)(uintptr_t)gp;
  v4u g0;
  g0[0] = 1u;                                              // count=1 (user D#)
  g0[1] = lds_addr;                                        // lds_addr[31:0]
  g0[2] = (unsigned)ga;                                    // global_addr lo
  g0[3] = (unsigned)((ga >> 32) & 0x1FFFFFFull) | (2u << 30);  // addr hi | type=2
  v8i g1;
  g1[0] = (int)0x00010000u;                                // data_size=1 -> 2B
  g1[1] = (int)((tensor_d0 & 0xffffu) << 16);              // tensor_dim0[15:0]
  g1[2] = (int)((tensor_d0 >> 16) | ((tensor_d1 & 0xffffu) << 16));
  g1[3] = (int)((tensor_d1 >> 16) | (tile_d0 << 16));      // | tile_dim0
  g1[4] = (int)tile_d1;                                    // tile_dim1 (dim2=0)
  g1[5] = (int)(unsigned)stride0;                          // dim0 stride lo32
  g1[6] = (int)(unsigned)(stride0 >> 32);                  // stride hi (dim1 stride=0)
  g1[7] = 0;
  v4i gz = {0, 0, 0, 0};
#if __clang_major__ >= 23
  v8i gz8 = {0, 0, 0, 0, 0, 0, 0, 0};
  __builtin_amdgcn_tensor_load_to_lds(g0, g1, gz, gz, gz8, 0);
#else
  __builtin_amdgcn_tensor_load_to_lds(g0, g1, gz, gz, 0);
#endif
}

// ---------------------------------------------------------------------------
// f32 -> f16 elementwise
// ---------------------------------------------------------------------------
__global__ void f32_to_f16_kernel(const float* __restrict__ in,
                                  half_t* __restrict__ out, int n) {
  int i = blockIdx.x * blockDim.x + threadIdx.x;
  int stride = gridDim.x * blockDim.x;
  for (; i < n; i += stride) out[i] = (half_t)in[i];
}

// Pack W[K,N] f32 row-major -> Wp[(k>>4)*N*16 + n*16 + (k&15)] f16
__global__ void pack_w_kernel(const float* __restrict__ in,
                              half_t* __restrict__ out, int K, int N) {
  int total = K * N;
  int i = blockIdx.x * blockDim.x + threadIdx.x;
  int stride = gridDim.x * blockDim.x;
  for (; i < total; i += stride) {
    int k = i / N, n = i - k * N;
    out[((size_t)(k >> 4) * N + n) * 16 + (k & 15)] = (half_t)in[i];
  }
}

// ---------------------------------------------------------------------------
// WMMA GEMM: C[M,N] f32 = A[M,K] f16 row-major * Bp (packed)
// 256 threads = 8 waves; wave -> 32x64 tile (2 M-tiles x 4 N-tiles, 8 accs);
// block -> 32 x 512.
// ---------------------------------------------------------------------------
__global__ __launch_bounds__(256)
void wmma_gemm(const half_t* __restrict__ A, const half_t* __restrict__ Bp,
               float* __restrict__ C, int M, int N, int K) {
  const int wave = threadIdx.x >> 5;
  const int lane = threadIdx.x & 31;
  const int lo = lane & 15, hi = lane >> 4;
  const int mblocks = M >> 5;
  const int m0 = (blockIdx.x % mblocks) << 5;
  const int n0 = (blockIdx.x / mblocks) * 512 + wave * 64;

  const half_t* arow0 = A + (size_t)(m0 + lo) * K + hi * 8;
  const half_t* arow1 = arow0 + (size_t)16 * K;
  const size_t bstride = (size_t)N * 16;
  const half_t* bbase = Bp + (size_t)(n0 + lo) * 16 + hi * 8;

  v8f acc[2][4];
#pragma unroll
  for (int mt = 0; mt < 2; ++mt)
#pragma unroll
    for (int a = 0; a < 4; ++a)
#pragma unroll
      for (int r = 0; r < 8; ++r) acc[mt][a][r] = 0.f;

  for (int k0 = 0; k0 < K; k0 += 32) {
    __builtin_prefetch(arow0 + k0 + 128, 0, 1);
    v16h a0 = combine16(*(const v8h*)(arow0 + k0),
                        *(const v8h*)(arow0 + k0 + 16));
    v16h a1 = combine16(*(const v8h*)(arow1 + k0),
                        *(const v8h*)(arow1 + k0 + 16));
    const half_t* bp = bbase + (size_t)(k0 >> 4) * bstride;
#pragma unroll
    for (int a = 0; a < 4; ++a) {
      v16h bf = combine16(*(const v8h*)(bp + a * 256),
                          *(const v8h*)(bp + a * 256 + bstride));
      acc[0][a] = wmma_f16(a0, bf, acc[0][a]);
      acc[1][a] = wmma_f16(a1, bf, acc[1][a]);
    }
  }

#pragma unroll
  for (int mt = 0; mt < 2; ++mt) {
    float* crow = C + (size_t)(m0 + mt * 16 + 8 * hi) * N + n0 + lo;
#pragma unroll
    for (int r = 0; r < 8; ++r)
#pragma unroll
      for (int a = 0; a < 4; ++a)
        crow[(size_t)r * N + a * 16] = acc[mt][a][r];
  }
}

// ---------------------------------------------------------------------------
// QK-LayerNorm + head split + V transpose.
// ---------------------------------------------------------------------------
__device__ __forceinline__ void ln_part(const float* __restrict__ src,
                                        const float* __restrict__ scale,
                                        half_t* __restrict__ dst,
                                        int b, int s, float extra,
                                        float* ssum, float* ssq) {
  const int tid = threadIdx.x;
  float4 xv = *(const float4*)(src + tid * 4);
  float sum = xv.x + xv.y + xv.z + xv.w;
  float sq  = xv.x * xv.x + xv.y * xv.y + xv.z * xv.z + xv.w * xv.w;
#pragma unroll
  for (int msk = 1; msk < 32; msk <<= 1) {
    sum += __shfl_xor(sum, msk);
    sq  += __shfl_xor(sq, msk);
  }
  if ((tid & 31) == 0) { ssum[tid >> 5] = sum; ssq[tid >> 5] = sq; }
  __syncthreads();
  float ts = 0.f, tq = 0.f;
#pragma unroll
  for (int i = 0; i < 8; ++i) { ts += ssum[i]; tq += ssq[i]; }
  float mean = ts * (1.f / 1024.f);
  float var  = tq * (1.f / 1024.f) - mean * mean;
  float rstd = __frsqrt_rn(var + 1e-6f);
  float vals[4] = {xv.x, xv.y, xv.z, xv.w};
#pragma unroll
  for (int i = 0; i < 4; ++i) {
    int d = tid * 4 + i, h = d >> 6, hd = d & 63;
    dst[(((size_t)(b * 16 + h)) * 2048 + s) * 64 + hd] =
        (half_t)((vals[i] - mean) * rstd * scale[d] * extra);
  }
  __syncthreads();
}

__global__ __launch_bounds__(256)
void ln_split_kernel(const float* __restrict__ qkv,
                     const float* __restrict__ q_scale,
                     const float* __restrict__ k_scale,
                     half_t* __restrict__ qh, half_t* __restrict__ kh,
                     half_t* __restrict__ vt) {
  __shared__ float ssum[8], ssq[8];
  const int row = blockIdx.x;          // b*2048 + s
  const int b = row >> 11, s = row & 2047;
  const float* base = qkv + (size_t)row * 3072;
  ln_part(base,        q_scale, qh, b, s, 0.125f, ssum, ssq);  // fold 1/sqrt(64)
  ln_part(base + 1024, k_scale, kh, b, s, 1.0f,   ssum, ssq);
  const int tid = threadIdx.x;
  float4 vv = *(const float4*)(base + 2048 + tid * 4);
  float vals[4] = {vv.x, vv.y, vv.z, vv.w};
#pragma unroll
  for (int i = 0; i < 4; ++i) {
    int d = tid * 4 + i, h = d >> 6, hd = d & 63;
    vt[(((size_t)(b * 16 + h)) * 64 + hd) * 2048 + s] = (half_t)vals[i];
  }
}

// ---------------------------------------------------------------------------
// Causal flash attention: one wave per (bh, 16-query-row tile).
// K/V tiles staged LDS via the Tensor Data Mover; fragments read with ds ops.
// ---------------------------------------------------------------------------
__global__ __launch_bounds__(32)
void attn_kernel(const half_t* __restrict__ qh, const half_t* __restrict__ kh,
                 const half_t* __restrict__ vt, half_t* __restrict__ ctxh) {
  constexpr int S = 2048;
  const int it = blockIdx.x & 127, bh = blockIdx.x >> 7;
  const int s0 = it << 4;
  const int lane = threadIdx.x & 31, lo = lane & 15, hi = lane >> 4;
  __shared__ _Float16 sk[32 * 64];   // K tile: [t_local][d]
  __shared__ _Float16 sv[64 * 32];   // V tile: [e][t_local]
  __shared__ _Float16 sp[16 * 32];   // probabilities
  const unsigned lds_k = (unsigned)(uintptr_t)(void*)sk;  // LDS offset = low 32 bits
  const unsigned lds_v = (unsigned)(uintptr_t)(void*)sv;

  // Q fragments (two K-halves of head dim 64), resident for the whole tile
  const half_t* qrow = qh + ((size_t)bh * S + s0 + lo) * 64 + hi * 16;
  v16h qf[2];
#pragma unroll
  for (int dh = 0; dh < 2; ++dh)
    qf[dh] = combine16(*(const v8h*)(qrow + dh * 32),
                       *(const v8h*)(qrow + dh * 32 + 8));

  float m[8], l[8];
  v8f ctx[4];
#pragma unroll
  for (int r = 0; r < 8; ++r) { m[r] = -1e30f; l[r] = 0.f; }
#pragma unroll
  for (int a = 0; a < 4; ++a)
#pragma unroll
    for (int r = 0; r < 8; ++r) ctx[a][r] = 0.f;

  const int nkb = (s0 + 47) >> 5;
  for (int kb = 0; kb < nkb; ++kb) {
    const int t0 = kb << 5;
    // WAR fence: all prior-iteration LDS reads must retire before TDM rewrites
    asm volatile("s_wait_dscnt 0x0" ::: "memory");
    // K tile: rows t0..t0+31, 64 halves each (row stride 64)
    tdm_load_2d(lds_k, kh + ((size_t)bh * S + t0) * 64,
                /*td0=*/64, /*td1=*/S, /*tile0=*/64, /*tile1=*/32,
                /*stride0=*/64ull);
    // V tile: 64 rows (e), 32 cols starting at t0 (row stride S)
    tdm_load_2d(lds_v, vt + ((size_t)bh * 64) * S + t0,
                /*td0=*/S, /*td1=*/64, /*tile0=*/32, /*tile1=*/64,
                /*stride0=*/(unsigned long long)S);
    __builtin_amdgcn_s_wait_tensorcnt(0);

    // ---- scores: q(16x64) . K^T -> 16x32 (K fragments from LDS)
    v8f sc0, sc1;
#pragma unroll
    for (int r = 0; r < 8; ++r) { sc0[r] = 0.f; sc1[r] = 0.f; }
#pragma unroll
    for (int dh = 0; dh < 2; ++dh) {
      const _Float16* kf0 = &sk[(lo)      * 64 + dh * 32 + hi * 16];
      const _Float16* kf1 = &sk[(16 + lo) * 64 + dh * 32 + hi * 16];
      sc0 = wmma_f16(qf[dh],
                     combine16(*(const v8h*)kf0, *(const v8h*)(kf0 + 8)), sc0);
      sc1 = wmma_f16(qf[dh],
                     combine16(*(const v8h*)kf1, *(const v8h*)(kf1 + 8)), sc1);
    }

    // ---- causal mask + online softmax
#pragma unroll
    for (int r = 0; r < 8; ++r) {
      const int rowg = s0 + r + 8 * hi;
      float a = sc0[r], b = sc1[r];
      if (t0 + lo > rowg)      a = -1e30f;
      if (t0 + 16 + lo > rowg) b = -1e30f;
      float mx = fmaxf(a, b);
#pragma unroll
      for (int msk = 1; msk < 16; msk <<= 1) mx = fmaxf(mx, __shfl_xor(mx, msk));
      float mnew = fmaxf(m[r], mx);
      float corr = __expf(m[r] - mnew);
      float p0 = __expf(a - mnew);
      float p1 = __expf(b - mnew);
      float rs = p0 + p1;
#pragma unroll
      for (int msk = 1; msk < 16; msk <<= 1) rs += __shfl_xor(rs, msk);
      l[r] = l[r] * corr + rs;
      m[r] = mnew;
#pragma unroll
      for (int a4 = 0; a4 < 4; ++a4) ctx[a4][r] *= corr;
      sp[(r + 8 * hi) * 32 + lo]      = (_Float16)p0;
      sp[(r + 8 * hi) * 32 + 16 + lo] = (_Float16)p1;
    }
    asm volatile("s_wait_dscnt 0x0" ::: "memory");

    // ---- reload P in A-fragment layout, V fragments from LDS
    v16h pf = combine16(*(const v8h*)(&sp[lo * 32 + hi * 8]),
                        *(const v8h*)(&sp[lo * 32 + hi * 8 + 16]));
#pragma unroll
    for (int a4 = 0; a4 < 4; ++a4) {
      const _Float16* vf = &sv[(a4 * 16 + lo) * 32 + hi * 8];
      ctx[a4] = wmma_f16(pf,
                         combine16(*(const v8h*)vf, *(const v8h*)(vf + 16)),
                         ctx[a4]);
    }
  }

  // ---- normalize + store f16 context (heads re-interleaved into d_model)
  const int b = bh >> 4, h = bh & 15;
#pragma unroll
  for (int r = 0; r < 8; ++r) {
    float inv = 1.0f / (l[r] + 1e-20f);
    half_t* crow =
        ctxh + ((size_t)(b * 2048 + s0 + r + 8 * hi)) * 1024 + h * 64 + lo;
#pragma unroll
    for (int a4 = 0; a4 < 4; ++a4) crow[a4 * 16] = (half_t)(ctx[a4][r] * inv);
  }
}

// ---------------------------------------------------------------------------
// launch
// ---------------------------------------------------------------------------
extern "C" void kernel_launch(void* const* d_in, const int* in_sizes, int n_in,
                              void* d_out, int out_size, void* d_ws, size_t ws_size,
                              hipStream_t stream) {
  const float* x     = (const float*)d_in[0];   // [2,2048,1024]
  const float* W_qkv = (const float*)d_in[1];   // [1024,3072]
  const float* q_s   = (const float*)d_in[2];
  const float* k_s   = (const float*)d_in[3];
  const float* W_out = (const float*)d_in[4];   // [1024,1024]
  float* out = (float*)d_out;

  char* w = (char*)d_ws;
  half_t* xh    = (half_t*)(w);                          //  8 MB
  half_t* wqkvp = (half_t*)(w + ((size_t)8  << 20));     //  6 MB
  half_t* woutp = (half_t*)(w + ((size_t)14 << 20));     //  2 MB
  float*  qkv   = (float*) (w + ((size_t)16 << 20));     // 48 MB
  half_t* qh    = (half_t*)(w + ((size_t)64 << 20));     //  8 MB
  half_t* kh    = (half_t*)(w + ((size_t)72 << 20));     //  8 MB
  half_t* vt    = (half_t*)(w + ((size_t)80 << 20));     //  8 MB
  half_t* ctxh  = (half_t*)(w + ((size_t)88 << 20));     //  8 MB

  f32_to_f16_kernel<<<4096, 256, 0, stream>>>(x, xh, 4096 * 1024);
  pack_w_kernel<<<6144, 256, 0, stream>>>(W_qkv, wqkvp, 1024, 3072);
  pack_w_kernel<<<2048, 256, 0, stream>>>(W_out, woutp, 1024, 1024);

  // qkv = x @ W_qkv : M=4096, N=3072 -> 128 m-blocks x 6 n-blocks
  wmma_gemm<<<768, 256, 0, stream>>>(xh, wqkvp, qkv, 4096, 3072, 1024);

  ln_split_kernel<<<4096, 256, 0, stream>>>(qkv, q_s, k_s, qh, kh, vt);

  attn_kernel<<<4096, 32, 0, stream>>>(qh, kh, vt, ctxh);

  // out = ctx @ W_out : M=4096, N=1024 -> 128 x 2
  wmma_gemm<<<256, 256, 0, stream>>>(ctxh, woutp, out, 4096, 1024, 1024);
}
